// PlayerPerformanceLSTM_53669911331243
// MI455X (gfx1250) — matrix operand order repaired and checked by reading, not compiled, not run on hardware
//
#include <hip/hip_runtime.h>
#include <hip/hip_bf16.h>

typedef __attribute__((ext_vector_type(16))) _Float16 v16h;
typedef __attribute__((ext_vector_type(8)))  _Float16 v8h;
typedef __attribute__((ext_vector_type(8)))  float    v8f;

#define B_TOT 4096
#define SEQ   512
#define FIN   11
#define HID   64
#define GATES 256
#define BT    16
#define NB1   1024

// Branch-free activations: v_exp_f32 + v_rcp_f32, no EXEC divergence.
__device__ __forceinline__ float fast_sigmoid(float x) {
  float e = __builtin_amdgcn_exp2f(-1.44269504f * x);   // exp(-x)
  return __builtin_amdgcn_rcpf(1.0f + e);
}
__device__ __forceinline__ float fast_tanh(float x) {
  float e = __builtin_amdgcn_exp2f(-2.88539008f * fabsf(x));  // exp(-2|x|)
  float t = (1.0f - e) * __builtin_amdgcn_rcpf(1.0f + e);
  return copysignf(t, x);
}

// ---------------- Stage 1: BN statistics (deterministic partials) ----------------
__global__ __launch_bounds__(256) void bn_stats1(const float* __restrict__ x,
                                                 float* __restrict__ partial) {
  const int ROWS = B_TOT * SEQ;
  int tid = threadIdx.x;
  int gtid = blockIdx.x * 256 + tid;
  int stride = gridDim.x * 256;
  float s[FIN], q[FIN];
#pragma unroll
  for (int f = 0; f < FIN; ++f) { s[f] = 0.f; q[f] = 0.f; }
  for (int r = gtid; r < ROWS; r += stride) {
    const float* p = x + (size_t)r * FIN;
#pragma unroll
    for (int f = 0; f < FIN; ++f) { float v = p[f]; s[f] += v; q[f] += v * v; }
  }
  __shared__ float red[2 * FIN][256];
#pragma unroll
  for (int f = 0; f < FIN; ++f) { red[f][tid] = s[f]; red[FIN + f][tid] = q[f]; }
  __syncthreads();
  if (tid < 2 * FIN) {
    float a = 0.f;
    for (int i = 0; i < 256; ++i) a += red[tid][i];   // fixed order -> deterministic
    partial[blockIdx.x * (2 * FIN) + tid] = a;
  }
}

// ----- Stage 2: finish BN stats, fold BN into layer-0 weights, pack all f16 B -----
__global__ __launch_bounds__(256) void bn_finish_prep(
    const float* __restrict__ partial,
    const float* __restrict__ gamma, const float* __restrict__ beta,
    const float* __restrict__ Wih0, const float* __restrict__ bih0, const float* __restrict__ bhh0,
    const float* __restrict__ Whh0, const float* __restrict__ Wih1, const float* __restrict__ Whh1,
    const float* __restrict__ bih1, const float* __restrict__ bhh1,
    _Float16* __restrict__ Wih0effH, float* __restrict__ bias0, float* __restrict__ bias1,
    _Float16* __restrict__ Whh0h, _Float16* __restrict__ Wih1h, _Float16* __restrict__ Whh1h) {
  int tid = threadIdx.x;
  __shared__ float tot[2 * FIN];
  __shared__ float sc[FIN], sh[FIN];
  if (tid < 2 * FIN) {
    float a = 0.f;
    for (int b = 0; b < NB1; ++b) a += partial[b * (2 * FIN) + tid];
    tot[tid] = a;
  }
  __syncthreads();
  if (tid < FIN) {
    const float N = (float)B_TOT * (float)SEQ;
    float mean = tot[tid] / N;
    float var  = tot[FIN + tid] / N - mean * mean;
    float s = gamma[tid] * rsqrtf(var + 1e-5f);
    sc[tid] = s;
    sh[tid] = beta[tid] - mean * s;
  }
  __syncthreads();
  int g = tid;  // 256 gate rows
  float b = bih0[g] + bhh0[g];
#pragma unroll
  for (int f = 0; f < FIN; ++f) {
    float w = Wih0[g * FIN + f];
    Wih0effH[g * 32 + f] = (_Float16)(w * sc[f]);  // BN scale folded; K padded to 32
    b += w * sh[f];                                // BN shift folded into bias
  }
#pragma unroll
  for (int f = FIN; f < 32; ++f) Wih0effH[g * 32 + f] = (_Float16)0.f;
  bias0[g] = b;
  bias1[g] = bih1[g] + bhh1[g];
  for (int k = 0; k < HID; ++k) {
    Whh0h[g * HID + k] = (_Float16)Whh0[g * HID + k];
    Wih1h[g * HID + k] = (_Float16)Wih1[g * HID + k];
    Whh1h[g * HID + k] = (_Float16)Whh1[g * HID + k];
  }
}

// A-fragment for v_wmma_f32_16x16x32_f16 from LDS matrix with `ldk` halfs per row:
// lane row m = lane&15; element e -> K = kbase + 16*(lane>>4) + e
__device__ __forceinline__ v16h load_afrag(const _Float16* hS, int ldk, int kbase, int m, int lhi) {
  union { v16h v; v8h p[2]; } u;
  const _Float16* p = hS + m * ldk + kbase + lhi * 16;
  u.p[0] = *(const v8h*)(p);
  u.p[1] = *(const v8h*)(p + 8);
  return u.v;
}

// ---------------- Fused persistent 2-layer LSTM + FC head ----------------
__global__ __launch_bounds__(256) void lstm_fused(
    const float* __restrict__ x,
    const _Float16* __restrict__ Wih0effH, const float* __restrict__ bias0g,
    const float* __restrict__ bias1g,
    const _Float16* __restrict__ Whh0h, const _Float16* __restrict__ Wih1h,
    const _Float16* __restrict__ Whh1h,
    const float* __restrict__ fc1W, const float* __restrict__ fc1b,
    const float* __restrict__ fc2W, const float* __restrict__ fc2b,
    float* __restrict__ out) {
  __shared__ float    gatesS[BT * GATES];   // 16 KB, shared by both layers
  __shared__ float    c1S[BT * HID];
  __shared__ float    c2S[BT * HID];
  __shared__ float    h2fS[BT * HID];
  __shared__ _Float16 h1hS[BT * HID];
  __shared__ _Float16 h2hS[BT * HID];
  __shared__ _Float16 xhS[BT * 32];         // f16 x, K zero-padded 11 -> 32
  __shared__ float    b0S[GATES];
  __shared__ float    b1S[GATES];
  __shared__ float    hidS[BT * 32];

  const int tid  = threadIdx.x;
  const int lane = tid & 31;
  const int wv   = tid >> 5;       // 8 waves, 2 N-tiles each
  const int colT = lane & 15;
  const int lhi  = lane >> 4;
  const int b0   = blockIdx.x * BT;

  if (tid < GATES) { b0S[tid] = bias0g[tid]; b1S[tid] = bias1g[tid]; }
#pragma unroll
  for (int k = 0; k < 4; ++k) {
    int u = tid * 4 + k;
    c1S[u] = 0.f; c2S[u] = 0.f;
    h1hS[u] = (_Float16)0.f; h2hS[u] = (_Float16)0.f;
  }
  xhS[tid] = (_Float16)0.f; xhS[tid + 256] = (_Float16)0.f;  // padding stays zero

  // B fragments: lane l, element e == row-major W[N=tile*16+(l&15)][kbase+16*(l>>4)+e]
  // -> single 32B contiguous loads; loaded ONCE, reused for all 512 steps.
  v16h bWx[2], bW0[2][2], bI1[2][2], bH1[2][2];
#pragma unroll
  for (int tile = 0; tile < 2; ++tile) {
    int n = (2 * wv + tile) * 16 + colT;
    bWx[tile] = *(const v16h*)(Wih0effH + n * 32 + lhi * 16);
#pragma unroll
    for (int kc = 0; kc < 2; ++kc) {
      int off = n * HID + kc * 32 + lhi * 16;
      bW0[tile][kc] = *(const v16h*)(Whh0h + off);
      bI1[tile][kc] = *(const v16h*)(Wih1h + off);
      bH1[tile][kc] = *(const v16h*)(Whh1h + off);
    }
  }

  // software-pipelined x load: xv holds x(t), prefetch x(t+1) right after B1
  const int xr = tid / FIN, xf = tid % FIN;
  float xv = 0.f;
  if (tid < BT * FIN) xv = x[((size_t)(b0 + xr) * SEQ + 0) * FIN + xf];
  __syncthreads();

  for (int t = 0; t < SEQ; ++t) {
    if (tid < BT * FIN) xhS[xr * 32 + xf] = (_Float16)xv;
    __syncthreads();  // B1: x_t ready; gatesS free for reuse
    if (tid < BT * FIN && t + 1 < SEQ)
      xv = x[((size_t)(b0 + xr) * SEQ + (t + 1)) * FIN + xf];  // hide latency over full step

    // ---- layer 0: gates = bias0 + x16 @ Wih0eff^T + h1 @ Whh0^T ----
    {
      v16h ax = load_afrag(xhS, 32, 0, colT, lhi);
      v16h a0 = load_afrag(h1hS, HID, 0, colT, lhi);
      v16h a1 = load_afrag(h1hS, HID, 32, colT, lhi);
#pragma unroll
      for (int tile = 0; tile < 2; ++tile) {
        int gcol = (2 * wv + tile) * 16 + colT;
        float bb = b0S[gcol];
        v8f acc;
#pragma unroll
        for (int j = 0; j < 8; ++j) acc[j] = bb;
        acc = __builtin_amdgcn_wmma_f32_16x16x32_f16(false, ax, false, bWx[tile],    (short)0, acc, false, false);
        acc = __builtin_amdgcn_wmma_f32_16x16x32_f16(false, a0, false, bW0[tile][0], (short)0, acc, false, false);
        acc = __builtin_amdgcn_wmma_f32_16x16x32_f16(false, a1, false, bW0[tile][1], (short)0, acc, false, false);
#pragma unroll
        for (int j = 0; j < 8; ++j) gatesS[(lhi * 8 + j) * GATES + gcol] = acc[j];
      }
    }
    __syncthreads();  // B2: gates0 ready

#pragma unroll
    for (int k = 0; k < 4; ++k) {            // layer-0 elementwise, 4 units/thread
      int u = tid * 4 + k;
      int r = u >> 6, j = u & 63;
      const float* gr = gatesS + r * GATES;
      float ig = fast_sigmoid(gr[j]);
      float fg = fast_sigmoid(gr[64 + j]);
      float gg = fast_tanh(gr[128 + j]);
      float og = fast_sigmoid(gr[192 + j]);
      float c = fg * c1S[u] + ig * gg;
      c1S[u] = c;
      h1hS[u] = (_Float16)(og * fast_tanh(c));
    }
    __syncthreads();  // B3: new h1 ready

    // ---- layer 1: gates = bias1 + h1 @ Wih1^T + h2 @ Whh1^T ----
    {
      v16h p0 = load_afrag(h1hS, HID, 0,  colT, lhi);
      v16h p1 = load_afrag(h1hS, HID, 32, colT, lhi);
      v16h q0 = load_afrag(h2hS, HID, 0,  colT, lhi);
      v16h q1 = load_afrag(h2hS, HID, 32, colT, lhi);
#pragma unroll
      for (int tile = 0; tile < 2; ++tile) {
        int gcol = (2 * wv + tile) * 16 + colT;
        float bb = b1S[gcol];
        v8f acc;
#pragma unroll
        for (int j = 0; j < 8; ++j) acc[j] = bb;
        acc = __builtin_amdgcn_wmma_f32_16x16x32_f16(false, p0, false, bI1[tile][0], (short)0, acc, false, false);
        acc = __builtin_amdgcn_wmma_f32_16x16x32_f16(false, p1, false, bI1[tile][1], (short)0, acc, false, false);
        acc = __builtin_amdgcn_wmma_f32_16x16x32_f16(false, q0, false, bH1[tile][0], (short)0, acc, false, false);
        acc = __builtin_amdgcn_wmma_f32_16x16x32_f16(false, q1, false, bH1[tile][1], (short)0, acc, false, false);
#pragma unroll
        for (int j = 0; j < 8; ++j) gatesS[(lhi * 8 + j) * GATES + gcol] = acc[j];
      }
    }
    __syncthreads();  // B4: gates1 ready

#pragma unroll
    for (int k = 0; k < 4; ++k) {            // layer-1 elementwise
      int u = tid * 4 + k;
      int r = u >> 6, j = u & 63;
      const float* gr = gatesS + r * GATES;
      float ig = fast_sigmoid(gr[j]);
      float fg = fast_sigmoid(gr[64 + j]);
      float gg = fast_tanh(gr[128 + j]);
      float og = fast_sigmoid(gr[192 + j]);
      float c = fg * c2S[u] + ig * gg;
      c2S[u] = c;
      float h = og * fast_tanh(c);
      h2hS[u] = (_Float16)h;
      if (t == SEQ - 1) h2fS[u] = h;         // uniform branch; only last step needed
    }
    // no barrier: next-iteration B1 covers gatesS/xhS reuse; B1..B3 cover h2hS
  }
  __syncthreads();

  // ---- FC head on h2[:, -1, :] ----
  for (int u = tid; u < BT * 32; u += 256) {
    int r = u >> 5, o = u & 31;
    float s = fc1b[o];
#pragma unroll 8
    for (int k = 0; k < HID; ++k) s += h2fS[r * HID + k] * fc1W[o * HID + k];
    hidS[u] = s > 0.f ? s : 0.f;
  }
  __syncthreads();
  if (tid < BT * 3) {
    int r = tid / 3, c = tid % 3;
    float s = fc2b[c];
#pragma unroll
    for (int k = 0; k < 32; ++k) s += hidS[r * 32 + k] * fc2W[c * 32 + k];
    out[(size_t)(b0 + r) * 3 + c] = s;
  }
}

extern "C" void kernel_launch(void* const* d_in, const int* in_sizes, int n_in,
                              void* d_out, int out_size, void* d_ws, size_t ws_size,
                              hipStream_t stream) {
  (void)in_sizes; (void)n_in; (void)out_size; (void)ws_size;
  const float* x        = (const float*)d_in[0];
  const float* bn_gamma = (const float*)d_in[1];
  const float* bn_beta  = (const float*)d_in[2];
  const float* Wih0     = (const float*)d_in[3];
  const float* Whh0     = (const float*)d_in[4];
  const float* bih0     = (const float*)d_in[5];
  const float* bhh0     = (const float*)d_in[6];
  const float* Wih1     = (const float*)d_in[7];
  const float* Whh1     = (const float*)d_in[8];
  const float* bih1     = (const float*)d_in[9];
  const float* bhh1     = (const float*)d_in[10];
  const float* fc1_W    = (const float*)d_in[11];
  const float* fc1_b    = (const float*)d_in[12];
  const float* fc2_W    = (const float*)d_in[13];
  const float* fc2_b    = (const float*)d_in[14];
  float* out = (float*)d_out;

  // workspace layout (bytes); everything fully overwritten each call
  char* ws = (char*)d_ws;
  float*    partial  = (float*)(ws + 0);          // 1024 * 22 * 4 = 90112
  _Float16* Wih0effH = (_Float16*)(ws + 90240);   // 256*32*2 = 16384 (32B aligned)
  float*    bias0    = (float*)(ws + 106624);     // 1024
  float*    bias1    = (float*)(ws + 107648);     // 1024
  _Float16* Whh0h    = (_Float16*)(ws + 108672);  // 32768 (32B aligned)
  _Float16* Wih1h    = (_Float16*)(ws + 141440);  // 32768
  _Float16* Whh1h    = (_Float16*)(ws + 174208);  // 32768 -> total 206976 B

  bn_stats1<<<NB1, 256, 0, stream>>>(x, partial);
  bn_finish_prep<<<1, 256, 0, stream>>>(partial, bn_gamma, bn_beta,
                                        Wih0, bih0, bhh0, Whh0, Wih1, Whh1, bih1, bhh1,
                                        Wih0effH, bias0, bias1, Whh0h, Wih1h, Whh1h);
  lstm_fused<<<B_TOT / BT, 256, 0, stream>>>(x, Wih0effH, bias0, bias1,
                                             Whh0h, Wih1h, Whh1h,
                                             fc1_W, fc1_b, fc2_W, fc2_b, out);
}
